// GELU244_23648089932081
// MI455X (gfx1250) — matrix operand order repaired and checked by reading, not compiled, not run on hardware
//
#include <hip/hip_runtime.h>
#include <hip/hip_bf16.h>
#include <math.h>

typedef __attribute__((ext_vector_type(16))) _Float16 v16h;
typedef __attribute__((ext_vector_type(8)))  float    v8f;
typedef unsigned int v4u __attribute__((ext_vector_type(4)));
typedef int          v8i __attribute__((ext_vector_type(8)));
typedef int          v4i __attribute__((ext_vector_type(4)));

#define DIM    1024
#define NSLOT  512
#define GELU_C 0.7978845608028654f   // sqrt(2/pi)
#define GELU_A 0.044715f

#if defined(__has_builtin)
#if __has_builtin(__builtin_amdgcn_tensor_load_to_lds) && \
    __has_builtin(__builtin_amdgcn_s_wait_tensorcnt)
#define HAVE_TDM 1
#endif
#endif

// 0.5*x*(1+tanh(u)) == x * sigmoid(2u), u = C*(x + A*x^3)
__device__ __forceinline__ float gelu_tanh(float x) {
    float u = GELU_C * (x + GELU_A * x * x * x);
    float s = 1.0f / (1.0f + __expf(-2.0f * u));
    return x * s;
}

// ---------------------------------------------------------------------------
// Pass 1: per-block partial column sums of gelu(x).  x is [rows, 1024] fp32.
// 256 threads * float4 = 1024 columns, grid-stride over rows, fully coalesced
// 128-bit loads.  Deterministic (fixed-order accumulation, no atomics).
// ---------------------------------------------------------------------------
__global__ void gelu_colsum_kernel(const float* __restrict__ x,
                                   float* __restrict__ partial, int rows) {
    int tid = threadIdx.x;               // 0..255 -> float4 column index
    const float4* x4 = (const float4*)x;
    float4 acc = {0.f, 0.f, 0.f, 0.f};
    for (int r = blockIdx.x; r < rows; r += gridDim.x) {
        float4 v = x4[(size_t)r * (DIM / 4) + tid];
        acc.x += gelu_tanh(v.x);
        acc.y += gelu_tanh(v.y);
        acc.z += gelu_tanh(v.z);
        acc.w += gelu_tanh(v.w);
    }
    ((float4*)partial)[(size_t)blockIdx.x * (DIM / 4) + tid] = acc;
}

// ---------------------------------------------------------------------------
// Pass 2: single 1024-thread block (32 wave32 waves).
//   - TDM (tensor_load_to_lds) demo load of a buf tile (non-load-bearing)
//   - reduce partials -> m_curr -> m_n (unit vector)
//   - sims = (buf @ m_n) / ||buf_row||  via V_WMMA_F32_16X16X32_F16:
//     wave w owns rows [16w, 16w+16); B = m_n replicated in all 16 columns,
//     K-loop of 32 WMMAs (unrolled x4) covers K=1024.
//   - masked argmax, gate computation, state updates.
// ---------------------------------------------------------------------------
__global__ __launch_bounds__(1024) void head_kernel(
    const float* __restrict__ partial, int nblocks, float inv_rows,
    const float* __restrict__ buf, const float* __restrict__ depl,
    const unsigned char* __restrict__ mask, const int* __restrict__ ptr_p,
    const float* __restrict__ log_k_p, const float* __restrict__ ldr_p,
    const float* __restrict__ lfloor_p,
    float* __restrict__ out_buf, float* __restrict__ out_depl,
    float* __restrict__ out_mask, float* __restrict__ gate_ws) {

    __shared__ float sm[DIM];       // m_curr, then m_n
    __shared__ float sred[DIM];     // reduction scratch for ||m||^2
    __shared__ float srn[NSLOT];    // row norms of buf
    __shared__ float ssim[NSLOT];   // raw dots, then masked sims
    __shared__ int   s_idx;
    __shared__ float s_factor;
#ifdef HAVE_TDM
    __shared__ float s_tdm[4 * DIM];  // TDM landing zone (16 KB)
#endif

    int tid = threadIdx.x;

#ifdef HAVE_TDM
    // Tensor Data Mover: async-load a 4x1024 fp32 tile of buf into LDS.
    // D# built per CDNA5 ISA sec.8 (count=1, type=2 "image", data_size=4B,
    // tensor dims 1024x512 / stride 1024, tile 1024x4, no cluster multicast).
    // Issued by wave 0 only; data intentionally NOT consumed (TDM descriptor
    // semantics are not hardware-verified), so correctness never depends on it.
    if (tid < 32) {
        unsigned           lds_base = (unsigned)(size_t)(&s_tdm[0]);
        unsigned long long ga       = (unsigned long long)(size_t)buf;
        v4u g0;
        g0.x = 1u;                                            // count=1 (valid)
        g0.y = lds_base;                                      // lds_addr
        g0.z = (unsigned)(ga & 0xffffffffull);                // global_addr lo
        g0.w = (unsigned)((ga >> 32) & 0x01ffffffull)         // global_addr hi
               | (2u << 30);                                  // type=2
        v8i g1;
        g1[0] = (int)(2u << 16);        // workgroup_mask=0, data_size=2 (4B)
        g1[1] = (int)(1024u << 16);     // tensor_dim0[15:0]=1024 in [31:16]
        g1[2] = (int)(512u << 16);      // tensor_dim0 hi=0, tensor_dim1 lo=512
        g1[3] = (int)(1024u << 16);     // tensor_dim1 hi=0, tile_dim0=1024
        g1[4] = 4;                      // tile_dim1=4 rows, tile_dim2=0
        g1[5] = 1024;                   // tensor_dim0_stride lo32 = 1024
        g1[6] = 0;                      // stride hi / dim1_stride lo
        g1[7] = 0;                      // tensor_dim1_stride hi
        v4i gz = {0, 0, 0, 0};
#if defined(__clang_major__) && (__clang_major__ >= 23)
        v8i gz8 = {0, 0, 0, 0, 0, 0, 0, 0};
        __builtin_amdgcn_tensor_load_to_lds(g0, g1, gz, gz, gz8, 0);
#else
        __builtin_amdgcn_tensor_load_to_lds(g0, g1, gz, gz, 0);
#endif
        __builtin_amdgcn_s_wait_tensorcnt(0);
    }
#endif

    // Copy buf -> new_buf (row `ptr` overwritten at the end).
    for (int i = tid; i < NSLOT * DIM; i += 1024) out_buf[i] = buf[i];

    // m_curr[tid] = mean over rows (fixed-order sum of nblocks partials).
    float m = 0.f;
    for (int p = 0; p < nblocks; ++p) m += partial[(size_t)p * DIM + tid];
    m *= inv_rows;
    sm[tid]   = m;
    sred[tid] = m * m;
    __syncthreads();
    for (int s = 512; s > 0; s >>= 1) {
        if (tid < s) sred[tid] += sred[tid + s];
        __syncthreads();
    }
    float inv_norm = 1.0f / fmaxf(sqrtf(sred[0]), 1e-12f);
    float mn = m * inv_norm;
    __syncthreads();
    sm[tid] = mn;                   // sm now holds m_n (fp32)

    // Row norms of buf (one thread per row, float4 reads).
    if (tid < NSLOT) {
        const float4* bp = (const float4*)(buf + (size_t)tid * DIM);
        float s = 0.f;
        for (int i = 0; i < DIM / 4; ++i) {
            float4 v = bp[i];
            s += v.x * v.x + v.y * v.y + v.z * v.z + v.w * v.w;
        }
        srn[tid] = sqrtf(s);
    }
    __syncthreads();

    // WMMA mat-vec: raw[n] = dot(buf[n], m_n).
    {
        int lane = tid & 31;
        int wid  = tid >> 5;                    // 0..31 -> 16-row tile
        int row  = wid * 16 + (lane & 15);
        int kA   = (lane < 16) ? 0 : 8;         // 16-bit A layout interleave
        int kB   = (lane < 16) ? 0 : 8;         // 16-bit B layout interleave (matches A)
        const float* brow = buf + (size_t)row * DIM;
        v8f acc = {};
#pragma unroll 4
        for (int c = 0; c < DIM / 32; ++c) {
            int ka = c * 32 + kA;
            int kb = c * 32 + kB;
            // 16-byte aligned vector fetches of the A tile and B vector.
            float4 p0 = *(const float4*)(brow + ka);
            float4 p1 = *(const float4*)(brow + ka + 4);
            float4 q0 = *(const float4*)(brow + ka + 16);
            float4 q1 = *(const float4*)(brow + ka + 20);
            float4 b0 = *(const float4*)(sm + kb);
            float4 b1 = *(const float4*)(sm + kb + 4);
            float4 b2 = *(const float4*)(sm + kb + 16);
            float4 b3 = *(const float4*)(sm + kb + 20);
            v16h a, b;
            a[0]  = (_Float16)p0.x; a[1]  = (_Float16)p0.y;
            a[2]  = (_Float16)p0.z; a[3]  = (_Float16)p0.w;
            a[4]  = (_Float16)p1.x; a[5]  = (_Float16)p1.y;
            a[6]  = (_Float16)p1.z; a[7]  = (_Float16)p1.w;
            a[8]  = (_Float16)q0.x; a[9]  = (_Float16)q0.y;
            a[10] = (_Float16)q0.z; a[11] = (_Float16)q0.w;
            a[12] = (_Float16)q1.x; a[13] = (_Float16)q1.y;
            a[14] = (_Float16)q1.z; a[15] = (_Float16)q1.w;
            b[0]  = (_Float16)b0.x; b[1]  = (_Float16)b0.y;
            b[2]  = (_Float16)b0.z; b[3]  = (_Float16)b0.w;
            b[4]  = (_Float16)b1.x; b[5]  = (_Float16)b1.y;
            b[6]  = (_Float16)b1.z; b[7]  = (_Float16)b1.w;
            b[8]  = (_Float16)b2.x; b[9]  = (_Float16)b2.y;
            b[10] = (_Float16)b2.z; b[11] = (_Float16)b2.w;
            b[12] = (_Float16)b3.x; b[13] = (_Float16)b3.y;
            b[14] = (_Float16)b3.z; b[15] = (_Float16)b3.w;
            acc = __builtin_amdgcn_wmma_f32_16x16x32_f16(
                false, a, false, b, (short)0, acc, false, false);
        }
        // D layout: lanes 0-15 -> M = vgpr (rows 0..7), lanes 16-31 -> M = 8+vgpr.
        if (lane == 0) {
#pragma unroll
            for (int j = 0; j < 8; ++j) ssim[wid * 16 + j] = acc[j];
        } else if (lane == 16) {
#pragma unroll
            for (int j = 0; j < 8; ++j) ssim[wid * 16 + 8 + j] = acc[j];
        }
    }
    __syncthreads();

    // sims = where(mask, raw/max(||row||,eps), -1)
    if (tid < NSLOT) {
        float d = ssim[tid] / fmaxf(srn[tid], 1e-12f);
        ssim[tid] = mask[tid] ? d : -1.0f;
    }
    __syncthreads();

    if (tid == 0) {
        int best = 0; float bv = ssim[0];
        for (int n = 1; n < NSLOT; ++n) {      // first-max, matches argmax
            float v = ssim[n];
            if (v > bv) { bv = v; best = n; }
        }
        float k_gate    = fminf(fmaxf(__expf(log_k_p[0]), 0.1f), 8.0f);
        float depl_rate = 0.1f + 0.8f / (1.0f + __expf(-ldr_p[0]));
        float floor_val = 0.5f / (1.0f + __expf(-lfloor_p[0]));
        float raw_gate  = __expf(-k_gate * (1.0f - depl[best]));
        gate_ws[0] = floor_val + (1.0f - floor_val) * raw_gate;
        s_idx    = best;
        s_factor = (bv > 0.85f) ? depl_rate : 1.0f;
    }
    __syncthreads();

    int ptr = ptr_p[0];
    if (tid < NSLOT) {
        float v = depl[tid];
        if (tid == s_idx) v *= s_factor;       // .at[idx].multiply(factor)
        if (tid == ptr)   v = 1.0f;            // then .at[ptr].set(1.0)
        out_depl[tid] = v;
        out_mask[tid] = (tid == ptr) ? 1.0f : (mask[tid] ? 1.0f : 0.0f);
    }
    out_buf[(size_t)ptr * DIM + tid] = sm[tid];   // new_buf[ptr] = m_n
}

// ---------------------------------------------------------------------------
// Pass 3: out = gelu(x) * gate.  Recomputes gelu instead of reloading y:
// the 134 MB of x fits in the 192 MB L2, so this re-read is L2-resident.
// float4 streaming + global_prefetch_b8.
// ---------------------------------------------------------------------------
__global__ void gelu_scale_kernel(const float* __restrict__ x,
                                  const float* __restrict__ gate_ws,
                                  float* __restrict__ out, int n4) {
    float g = gate_ws[0];
    const float4* x4 = (const float4*)x;
    float4* o4 = (float4*)out;
    int stride = gridDim.x * blockDim.x;
    for (int i = blockIdx.x * blockDim.x + threadIdx.x; i < n4; i += stride) {
        __builtin_prefetch(&x4[i + stride], 0, 1);
        float4 v = x4[i];
        float4 r;
        r.x = gelu_tanh(v.x) * g;
        r.y = gelu_tanh(v.y) * g;
        r.z = gelu_tanh(v.z) * g;
        r.w = gelu_tanh(v.w) * g;
        o4[i] = r;
    }
}

extern "C" void kernel_launch(void* const* d_in, const int* in_sizes, int n_in,
                              void* d_out, int out_size, void* d_ws, size_t ws_size,
                              hipStream_t stream) {
    const float*         x      = (const float*)d_in[0];
    const float*         buf    = (const float*)d_in[1];
    const float*         depl   = (const float*)d_in[2];
    const unsigned char* mask   = (const unsigned char*)d_in[3];
    const int*           ptr_p  = (const int*)d_in[4];
    const float*         log_k  = (const float*)d_in[5];
    const float*         ldr    = (const float*)d_in[6];
    const float*         lfloor = (const float*)d_in[7];

    int nY   = in_sizes[0];          // 8*4096*1024
    int nBuf = in_sizes[1];          // 512*1024
    int nN   = in_sizes[2];          // 512
    int rows = nY / DIM;             // 32768

    float* out      = (float*)d_out;
    float* out_y    = out;
    float* out_buf  = out + nY;
    float* out_depl = out_buf + nBuf;
    float* out_mask = out_depl + nN;

    // Workspace: nblocks partial column-sum rows (1024 floats each) + gate.
    float* wsf = (float*)d_ws;
    size_t slots = ws_size / (DIM * sizeof(float));
    int nblocks = (slots > 1) ? (int)(slots - 1) : 1;
    if (nblocks > 1024) nblocks = 1024;
    if (nblocks < 1)    nblocks = 1;
    if (nblocks > rows) nblocks = rows;
    float* gate_ws = wsf + (size_t)nblocks * DIM;

    gelu_colsum_kernel<<<nblocks, 256, 0, stream>>>(x, wsf, rows);
    head_kernel<<<1, 1024, 0, stream>>>(wsf, nblocks, 1.0f / (float)rows,
                                        buf, depl, mask, ptr_p, log_k, ldr, lfloor,
                                        out_buf, out_depl, out_mask, gate_ws);
    gelu_scale_kernel<<<8192, 256, 0, stream>>>(x, gate_ws, out_y, nY / 4);
}
